// ConditionedHamiltonianTransition_30305289240759
// MI455X (gfx1250) — matrix operand order repaired and checked by reading, not compiled, not run on hardware
//
#include <hip/hip_runtime.h>
#include <hip/hip_bf16.h>

// ---------------------------------------------------------------------------
// ConditionedHamiltonianTransition on MI455X (gfx1250, wave32, WMMA)
//
// B=65536, DYN=128, STAT=64, HALF=64, U=16, NOBS=20, H1=H2=512, NEULER=4
//
// Fused persistent-tile design: each 256-thread block (8 waves) owns a 64-row
// batch tile, keeps z in fp32 registers across all 4 Euler steps, activations
// in LDS as bf16, and does every GEMM with v_wmma_f32_16x16x32_bf16.
// Weights are converted once per launch to bf16 (plus transposed copies) so
// all WMMA B-operands stream contiguously from L2.
// ---------------------------------------------------------------------------

typedef __attribute__((ext_vector_type(16))) __bf16        v16bf;
typedef __attribute__((ext_vector_type(8)))  float         v8f;
typedef __attribute__((ext_vector_type(4)))  unsigned int  v4u;

#define TB     256
#define BM     64
#define HID    512
#define ZINc   192
#define DYNc   128
#define STATc  64
#define HALFc  64
#define Ucnt   16
#define NOBSc  20
#define BTOT   65536

// LDS layout (bytes)
#define ZF_OFF   0                          // bf16 [64][192]  = 24576
#define G_OFF    (BM * ZINc * 2)            // bf16 [64][512]  = 65536
#define H1_OFF   (G_OFF + BM * HID * 2)     // bf16 [64][512]  = 65536 (aliased by fp32 dz[64][128])
#define UT_OFF   (H1_OFF + BM * HID * 2)    // f32  [64][16]   = 4096
#define SMEM_SZ  (UT_OFF + BM * Ucnt * 4)   // 159744 B  (< 320 KB/WGP)

static __device__ __forceinline__ unsigned short f2bf(float f) {
  unsigned u = __float_as_uint(f);
  u += 0x7FFFu + ((u >> 16) & 1u);          // round-to-nearest-even
  return (unsigned short)(u >> 16);
}
static __device__ __forceinline__ float bf2f(unsigned short h) {
  return __uint_as_float(((unsigned)h) << 16);
}

union FragU { v16bf v; v4u q[2]; };

// A-fragment (16x32 bf16) per ISA 7.12.2: lane<16 row M=lane holds K={kb..kb+7,
// kb+16..kb+23}; lane>=16 same row, kb+=8.  Two contiguous 16B runs.
static __device__ __forceinline__ v16bf loadA(const unsigned short* buf, int ld,
                                              int m0, int k0, int lane) {
  int m  = m0 + (lane & 15);
  int kb = k0 + ((lane >> 4) << 3);
  const unsigned short* p = buf + m * ld + kb;
  FragU f;
  f.q[0] = *(const v4u*)(p);
  f.q[1] = *(const v4u*)(p + 16);
  return f.v;
}

// B-fragment (32x16 bf16): lane<16 col N=lane holds K={k0..k0+7, k0+16..k0+23};
// lane>=16 same col, k0+=8. W is row-major [N][K]: two contiguous 16B runs.
static __device__ __forceinline__ v16bf loadB(const unsigned short* W, int ld,
                                              int n0, int k0, int lane) {
  int n  = n0 + (lane & 15);
  int kb = k0 + ((lane >> 4) << 3);
  const unsigned short* p = W + n * ld + kb;
  FragU f;
  f.q[0] = *(const v4u*)(p);
  f.q[1] = *(const v4u*)(p + 16);
  return f.v;
}

static __device__ __forceinline__ v8f wmma_bf16(v16bf a, v16bf b, v8f c) {
  return __builtin_amdgcn_wmma_f32_16x16x32_bf16(false, a, false, b,
                                                 (short)0, c, false, false);
}

// 64x64 output patch per wave: 4 m-tiles x 4 n-tiles, K swept in 32-steps.
template <int KS>
static __device__ __forceinline__ void gemm4x4(const unsigned short* act, int lda,
                                               const unsigned short* W, int ldw,
                                               int wave, int lane, v8f acc[4][4]) {
  const v8f vz = {0.f, 0.f, 0.f, 0.f, 0.f, 0.f, 0.f, 0.f};
#pragma unroll
  for (int i = 0; i < 4; i++)
#pragma unroll
    for (int j = 0; j < 4; j++) acc[i][j] = vz;

#pragma unroll 2
  for (int ks = 0; ks < KS; ks++) {
    const int k0 = ks << 5;
    v16bf a[4], b[4];
#pragma unroll
    for (int i = 0; i < 4; i++) a[i] = loadA(act, lda, 16 * i, k0, lane);
#pragma unroll
    for (int j = 0; j < 4; j++) b[j] = loadB(W, ldw, wave * 64 + 16 * j, k0, lane);
#pragma unroll
    for (int i = 0; i < 4; i++)
#pragma unroll
      for (int j = 0; j < 4; j++) acc[i][j] = wmma_bf16(a[i], b[j], acc[i][j]);
  }
}

// ---------------------------------------------------------------------------
// Prep: fp32 -> bf16 weight copies (+ transposes) into workspace.
//   W1n [512][192] = W1            (GEMM1 B)
//   W1Tn[128][512] : W1Tn[c][j] = W1[j][c]   (GEMM4 B, only first 128 cols)
//   W2n [512][512] = W2            (GEMM2 B)
//   W2Tn[512][512] : W2Tn[a][b] = W2[b][a]   (GEMM3 B)
// ---------------------------------------------------------------------------
__global__ __launch_bounds__(256) void hnn_prep_kernel(
    const float* __restrict__ W1, const float* __restrict__ W2,
    unsigned short* __restrict__ W1n, unsigned short* __restrict__ W1Tn,
    unsigned short* __restrict__ W2n, unsigned short* __restrict__ W2Tn) {
  int idx = blockIdx.x * 256 + threadIdx.x;             // grid covers 512*512
  if (idx < 512 * ZINc) W1n[idx] = f2bf(W1[idx]);
  if (idx < DYNc * HID) {
    int c = idx >> 9, j = idx & 511;
    W1Tn[idx] = f2bf(W1[j * ZINc + c]);
  }
  if (idx < HID * HID) {
    W2n[idx] = f2bf(W2[idx]);
    int a = idx >> 9, b = idx & 511;
    W2Tn[idx] = f2bf(W2[b * HID + a]);
  }
}

// ---------------------------------------------------------------------------
// Main fused kernel: 4 Euler steps + observation epilogue.
// ---------------------------------------------------------------------------
__global__ __launch_bounds__(TB) void hnn_step_kernel(
    const float* __restrict__ z_dyn, const float* __restrict__ z_static,
    const float* __restrict__ dtp, const float* __restrict__ ut,
    const float* __restrict__ b1, const float* __restrict__ b2,
    const float* __restrict__ W3, const float* __restrict__ B_ctrl,
    const float* __restrict__ Cm, const float* __restrict__ Dm,
    const unsigned short* __restrict__ W1n, const unsigned short* __restrict__ W1Tn,
    const unsigned short* __restrict__ W2n, const unsigned short* __restrict__ W2Tn,
    float* __restrict__ out_z, float* __restrict__ out_y) {
  extern __shared__ char smem[];
  unsigned short* zf  = (unsigned short*)(smem + ZF_OFF);   // [64][192] bf16
  unsigned short* gbf = (unsigned short*)(smem + G_OFF);    // [64][512] bf16
  unsigned short* h1b = (unsigned short*)(smem + H1_OFF);   // [64][512] bf16
  float*          dzb = (float*)(smem + H1_OFF);            // [64][128] f32 (aliases h1b)
  float*          utb = (float*)(smem + UT_OFF);            // [64][16]  f32

  const int tid  = threadIdx.x;
  const int lane = tid & 31;
  const int wave = tid >> 5;
  const int row0 = blockIdx.x * BM;

  const float dtv   = dtp[0];
  const float hstep = dtv * 0.25f;                          // dt / NEULER

  // ---- stage ut tile and z_static into LDS ----
  for (int idx = tid; idx < BM * Ucnt; idx += TB)
    utb[idx] = ut[(size_t)row0 * Ucnt + idx];
  for (int idx = tid; idx < BM * STATc; idx += TB) {
    int m = idx >> 6, c = idx & 63;
    zf[m * ZINc + DYNc + c] = f2bf(z_static[(size_t)(row0 + m) * STATc + c]);
  }

  // ---- z held in registers: thread owns row zm, 32 contiguous cols ----
  const int zm    = tid >> 2;
  const int cbase = (tid & 3) * 32;
  float zreg[32];
#pragma unroll
  for (int i = 0; i < 32; i++) {
    float v = z_dyn[(size_t)(row0 + zm) * DYNc + cbase + i];
    zreg[i] = v;
    zf[zm * ZINc + cbase + i] = f2bf(v);
  }
  __syncthreads();

  // ---- step-invariant control term (ut @ B_ctrl^T) for owned cols >= 64 ----
  float ubc[32];
  if (cbase >= HALFc) {
#pragma unroll 4
    for (int i = 0; i < 32; i++) {
      int c2 = cbase - HALFc + i;
      float s = 0.f;
#pragma unroll
      for (int u = 0; u < Ucnt; u++) s += utb[zm * Ucnt + u] * B_ctrl[c2 * Ucnt + u];
      ubc[i] = s;
    }
  }

  for (int step = 0; step < 4; ++step) {
    v8f acc[4][4];

    // ===== GEMM1: pre1 = zf @ W1^T (K=192) ; h1 = softplus(pre1+b1) =====
    gemm4x4<6>(zf, ZINc, W1n, ZINc, wave, lane, acc);
    {
      const int mh = (lane >> 4) << 3;
#pragma unroll
      for (int j = 0; j < 4; j++) {
        const int n = wave * 64 + j * 16 + (lane & 15);
        const float bias = b1[n];
#pragma unroll
        for (int i = 0; i < 4; i++)
#pragma unroll
          for (int r = 0; r < 8; r++) {
            const int m = i * 16 + mh + r;
            float x  = acc[i][j][r] + bias;
            float sp = (x > 15.f) ? x : __logf(1.f + __expf(x));
            h1b[m * HID + n] = f2bf(sp);
          }
      }
    }
    __syncthreads();

    // ===== GEMM2: pre2 = h1 @ W2^T (K=512) ; g2 = W3 * sigmoid(pre2+b2) =====
    gemm4x4<16>(h1b, HID, W2n, HID, wave, lane, acc);
    {
      const int mh = (lane >> 4) << 3;
#pragma unroll
      for (int j = 0; j < 4; j++) {
        const int n = wave * 64 + j * 16 + (lane & 15);
        const float bias = b2[n];
        const float w3n  = W3[n];
#pragma unroll
        for (int i = 0; i < 4; i++)
#pragma unroll
          for (int r = 0; r < 8; r++) {
            const int m = i * 16 + mh + r;
            float x  = acc[i][j][r] + bias;
            float s2 = 1.f / (1.f + __expf(-x));
            gbf[m * HID + n] = f2bf(w3n * s2);
          }
      }
    }
    __syncthreads();

    // ===== GEMM3: g_h1 = g2 @ W2 (via W2T, K=512) ;
    //       g_pre1 = g_h1 * sigmoid(pre1) with sigmoid(pre1)=1-exp(-h1) =====
    gemm4x4<16>(gbf, HID, W2Tn, HID, wave, lane, acc);
    __syncthreads();   // everyone done READING g2 before in-place overwrite
    {
      const int mh = (lane >> 4) << 3;
#pragma unroll
      for (int j = 0; j < 4; j++) {
        const int n = wave * 64 + j * 16 + (lane & 15);
#pragma unroll
        for (int i = 0; i < 4; i++)
#pragma unroll
          for (int r = 0; r < 8; r++) {
            const int m = i * 16 + mh + r;
            float s1 = 1.f - __expf(-bf2f(h1b[m * HID + n]));
            gbf[m * HID + n] = f2bf(acc[i][j][r] * s1);
          }
      }
    }
    __syncthreads();

    // ===== GEMM4: dzf = g_pre1 @ W1[:, :128] (via W1T, K=512) -> dz f32 =====
    {
      const v8f vz = {0.f, 0.f, 0.f, 0.f, 0.f, 0.f, 0.f, 0.f};
      v8f a4[4] = {vz, vz, vz, vz};
#pragma unroll 2
      for (int ks = 0; ks < 16; ks++) {
        const int k0 = ks << 5;
        v16bf a[4];
#pragma unroll
        for (int i = 0; i < 4; i++) a[i] = loadA(gbf, HID, 16 * i, k0, lane);
        v16bf bfr = loadB(W1Tn, HID, wave * 16, k0, lane);
#pragma unroll
        for (int i = 0; i < 4; i++) a4[i] = wmma_bf16(a[i], bfr, a4[i]);
      }
      const int n  = wave * 16 + (lane & 15);
      const int mh = (lane >> 4) << 3;
#pragma unroll
      for (int i = 0; i < 4; i++)
#pragma unroll
        for (int r = 0; r < 8; r++)
          dzb[(i * 16 + mh + r) * DYNc + n] = a4[i][r];   // h1b dead; alias OK
    }
    __syncthreads();

    // ===== symplectic-Euler style update: z += h * [dHdp ; -dHdq + u B^T] ====
#pragma unroll
    for (int i = 0; i < 32; i++) {
      const int c = cbase + i;
      float rhs;
      if (c < HALFc) rhs = dzb[zm * DYNc + c + HALFc];
      else           rhs = -dzb[zm * DYNc + c - HALFc] + ubc[i];
      zreg[i] += hstep * rhs;
      zf[zm * ZINc + c] = f2bf(zreg[i]);
    }
    __syncthreads();
  }

  // ---- outputs: z (fp32), then yt = z@C^T + dt*(ut@D^T) ----
#pragma unroll
  for (int i = 0; i < 32; i++) {
    out_z[(size_t)(row0 + zm) * DYNc + cbase + i] = zreg[i];
    dzb[zm * DYNc + cbase + i] = zreg[i];   // stash fp32 z in LDS for yt
  }
  __syncthreads();
  for (int idx = tid; idx < BM * NOBSc; idx += TB) {
    const int m = idx / NOBSc, o = idx % NOBSc;
    float s = 0.f;
#pragma unroll 4
    for (int k = 0; k < DYNc; k++) s += dzb[m * DYNc + k] * Cm[o * DYNc + k];
    float su = 0.f;
#pragma unroll
    for (int u = 0; u < Ucnt; u++) su += utb[m * Ucnt + u] * Dm[o * Ucnt + u];
    out_y[(size_t)(row0 + m) * NOBSc + o] = s + dtv * su;
  }
}

// ---------------------------------------------------------------------------
extern "C" void kernel_launch(void* const* d_in, const int* in_sizes, int n_in,
                              void* d_out, int out_size, void* d_ws, size_t ws_size,
                              hipStream_t stream) {
  (void)in_sizes; (void)n_in; (void)out_size; (void)ws_size;

  const float* z_dyn  = (const float*)d_in[0];
  const float* z_stat = (const float*)d_in[1];
  const float* dtp    = (const float*)d_in[2];
  const float* utp    = (const float*)d_in[3];
  const float* W1     = (const float*)d_in[4];
  const float* b1     = (const float*)d_in[5];
  const float* W2     = (const float*)d_in[6];
  const float* b2     = (const float*)d_in[7];
  const float* W3     = (const float*)d_in[8];
  // d_in[9] = b3: unused (does not affect grad_H, and yt has no b3 term)
  const float* Bctrl  = (const float*)d_in[10];
  const float* Cm     = (const float*)d_in[11];
  const float* Dm     = (const float*)d_in[12];

  char* ws = (char*)d_ws;                                  // ~1.4 MB used
  unsigned short* W1n  = (unsigned short*)(ws);
  unsigned short* W1Tn = (unsigned short*)(ws + 512 * 192 * 2);
  unsigned short* W2n  = (unsigned short*)(ws + 512 * 192 * 2 + 128 * 512 * 2);
  unsigned short* W2Tn = (unsigned short*)(ws + 512 * 192 * 2 + 128 * 512 * 2 + 512 * 512 * 2);

  float* out_z = (float*)d_out;
  float* out_y = out_z + (size_t)BTOT * DYNc;

  hnn_prep_kernel<<<(512 * 512) / 256, 256, 0, stream>>>(W1, W2, W1n, W1Tn, W2n, W2Tn);
  hnn_step_kernel<<<BTOT / BM, TB, SMEM_SZ, stream>>>(
      z_dyn, z_stat, dtp, utp, b1, b2, W3, Bctrl, Cm, Dm,
      W1n, W1Tn, W2n, W2Tn, out_z, out_y);
}
